// LiftedLoss_37812892074052
// MI455X (gfx1250) — matrix operand order repaired and checked by reading, not compiled, not run on hardware
//
#include <hip/hip_runtime.h>
#include <hip/hip_bf16.h>

typedef __attribute__((ext_vector_type(16))) _Float16 v16h;
typedef __attribute__((ext_vector_type(8)))  _Float16 v8h;
typedef __attribute__((ext_vector_type(4)))  _Float16 v4h;
typedef __attribute__((ext_vector_type(8)))  float    v8f;
typedef __attribute__((ext_vector_type(4)))  float    v4f;

#define B_ROWS 4096
#define DIMK   128
#define NTILE  (B_ROWS / 16)   // 256
#define MARGIN 1.0f

// Single-instruction v_sqrt_f32 (no IEEE refinement sequence).
__device__ inline float fast_sqrt(float x) { return __builtin_amdgcn_sqrtf(x); }

// ---- WMMA fragment loaders (16-bit A 16x32 and B 32x16 ISA layouts) ----
// A: lane<16 holds M=lane, halfs[0..7]=K kb+0..7, halfs[8..15]=K kb+16..23
//    lane>=16 same M, K offsets +8.  Two 16B loads at kb+8*hi and kb+16+8*hi.
__device__ inline v16h load_fragA(const _Float16* row, int kb, int laneHi) {
    const v8h* p = reinterpret_cast<const v8h*>(row + kb + laneHi * 8);
    union { v16h v; v8h h[2]; } u;
    u.h[0] = p[0];
    u.h[1] = p[2];   // +16 halfs from first chunk
    return u.v;
}
// B: lane<16 holds N=lane, K = kb+0..7, kb+16..23 ; lane>=16 K offsets +8 (same order as A).
__device__ inline v16h load_fragB(const _Float16* row, int kb, int laneHi) {
    const v8h* p = reinterpret_cast<const v8h*>(row + kb + laneHi * 8);
    union { v16h v; v8h h[2]; } u;
    u.h[0] = p[0];
    u.h[1] = p[2];
    return u.v;
}

__device__ inline v8f wmma_acc(v16h a, v16h b, v8f c) {
    return __builtin_amdgcn_wmma_f32_16x16x32_f16(
        /*neg_a=*/false, a, /*neg_b=*/false, b,
        /*c_mod=*/(short)0, c, /*reuse_a=*/false, /*reuse_b=*/false);
}

// ---- Kernel 1: f32->f16 copy, row squared norms, zero accumulators ----
__global__ __launch_bounds__(256) void prep_kernel(const float* __restrict__ score,
                                                   _Float16* __restrict__ Xh,
                                                   float* __restrict__ mag,
                                                   float* __restrict__ neg_sum,
                                                   float* __restrict__ loss,
                                                   float* __restrict__ cnt) {
    const int tid  = threadIdx.x;
    const int lane = tid & 31;
    const int wave = tid >> 5;
    const int row  = blockIdx.x * 8 + wave;           // grid.x = 512 -> 4096 rows

    const float* src = score + (size_t)row * DIMK + lane * 4;
    v4f x = *reinterpret_cast<const v4f*>(src);
    float ss = x.x * x.x + x.y * x.y + x.z * x.z + x.w * x.w;

    v4h h;
    h.x = (_Float16)x.x; h.y = (_Float16)x.y; h.z = (_Float16)x.z; h.w = (_Float16)x.w;
    *reinterpret_cast<v4h*>(Xh + (size_t)row * DIMK + lane * 4) = h;

    ss += __shfl_xor(ss, 1, 32);
    ss += __shfl_xor(ss, 2, 32);
    ss += __shfl_xor(ss, 4, 32);
    ss += __shfl_xor(ss, 8, 32);
    ss += __shfl_xor(ss, 16, 32);
    if (lane == 0) mag[row] = ss;

    const int gid = blockIdx.x * 256 + tid;
    if (gid < B_ROWS) neg_sum[gid] = 0.0f;
    if (gid == 0) { *loss = 0.0f; *cnt = 0.0f; }
}

// ---- Kernel 2: neg_sum[i] = sum_{j: t_j != t_i} exp(margin - d_ij) ----
__global__ __launch_bounds__(256) void negsum_kernel(const _Float16* __restrict__ Xh,
                                                     const float* __restrict__ mag,
                                                     const long long* __restrict__ tgt,
                                                     float* __restrict__ neg_sum) {
    const int lane   = threadIdx.x & 31;
    const int wave   = threadIdx.x >> 5;
    const int laneLo = lane & 15;
    const int laneHi = lane >> 4;
    const int It     = blockIdx.x;                    // grid.x = 256 row strips

    // A fragments for this 16-row strip: constant across all column tiles.
    const _Float16* arow = Xh + (size_t)(It * 16 + laneLo) * DIMK;
    v16h a0 = load_fragA(arow, 0,  laneHi);
    v16h a1 = load_fragA(arow, 32, laneHi);
    v16h a2 = load_fragA(arow, 64, laneHi);
    v16h a3 = load_fragA(arow, 96, laneHi);

    const int ibase = It * 16 + laneHi * 8;           // C-fragment row base for this lane
    float magI[8]; int tI[8];
#pragma unroll
    for (int v = 0; v < 8; ++v) {
        magI[v] = mag[ibase + v];
        tI[v]   = (int)tgt[ibase + v];
    }

    float acc[8];
#pragma unroll
    for (int v = 0; v < 8; ++v) acc[v] = 0.0f;

    for (int Jt = wave; Jt < NTILE; Jt += 8) {
        const int jc = Jt * 16 + laneLo;              // C-fragment column for this lane
        const float magJ = mag[jc];
        const int   tJ   = (int)tgt[jc];
        const _Float16* brow = Xh + (size_t)jc * DIMK;

        v8f c = {};
        c = wmma_acc(a0, load_fragB(brow, 0,  laneHi), c);
        c = wmma_acc(a1, load_fragB(brow, 32, laneHi), c);
        c = wmma_acc(a2, load_fragB(brow, 64, laneHi), c);
        c = wmma_acc(a3, load_fragB(brow, 96, laneHi), c);

        // Branchless: one v_sqrt + one v_exp per element, cndmask'ed addend.
#pragma unroll
        for (int v = 0; v < 8; ++v) {
            float d2   = magI[v] + magJ - 2.0f * c[v];
            float dist = d2 > 0.0f ? fast_sqrt(d2) : 0.0f;
            float e    = __expf(MARGIN - dist);
            acc[v]    += (tI[v] != tJ) ? e : 0.0f;
        }
    }

    // Row sums live across the 16 lanes of each half-wave.
#pragma unroll
    for (int v = 0; v < 8; ++v) {
        float s = acc[v];
        s += __shfl_xor(s, 1, 32);
        s += __shfl_xor(s, 2, 32);
        s += __shfl_xor(s, 4, 32);
        s += __shfl_xor(s, 8, 32);
        acc[v] = s;
    }
    if (laneLo == 0) {
#pragma unroll
        for (int v = 0; v < 8; ++v) atomicAdd(&neg_sum[ibase + v], acc[v]);
    }
}

// ---- Kernel 3: sum over same-class i<j of relu(log(ns_i+ns_j)+d_ij)^2 ----
__global__ __launch_bounds__(32) void pairloss_kernel(const _Float16* __restrict__ Xh,
                                                      const float* __restrict__ mag,
                                                      const long long* __restrict__ tgt,
                                                      const float* __restrict__ neg_sum,
                                                      float* __restrict__ loss,
                                                      float* __restrict__ cnt) {
    const int Jt = blockIdx.x;
    const int It = blockIdx.y;
    if (Jt < It) return;                              // uniform per wave: EXEC stays all-ones

    const int lane   = threadIdx.x & 31;
    const int laneLo = lane & 15;
    const int laneHi = lane >> 4;

    const _Float16* arow = Xh + (size_t)(It * 16 + laneLo) * DIMK;
    const int jc = Jt * 16 + laneLo;
    const _Float16* brow = Xh + (size_t)jc * DIMK;

    v8f c = {};
    c = wmma_acc(load_fragA(arow, 0,  laneHi), load_fragB(brow, 0,  laneHi), c);
    c = wmma_acc(load_fragA(arow, 32, laneHi), load_fragB(brow, 32, laneHi), c);
    c = wmma_acc(load_fragA(arow, 64, laneHi), load_fragB(brow, 64, laneHi), c);
    c = wmma_acc(load_fragA(arow, 96, laneHi), load_fragB(brow, 96, laneHi), c);

    const int ibase = It * 16 + laneHi * 8;
    const float magJ = mag[jc];
    const int   tJ   = (int)tgt[jc];
    const float nsJ  = neg_sum[jc];

    float magI[8], nsI[8]; int tI[8];
#pragma unroll
    for (int v = 0; v < 8; ++v) {
        magI[v] = mag[ibase + v];
        nsI[v]  = neg_sum[ibase + v];
        tI[v]   = (int)tgt[ibase + v];
    }

    float accL = 0.0f, accC = 0.0f;
#pragma unroll
    for (int v = 0; v < 8; ++v) {
        const int gi = ibase + v;
        const bool m = (tI[v] == tJ) && (gi < jc);
        float d2   = magI[v] + magJ - 2.0f * c[v];
        float dist = d2 > 0.0f ? fast_sqrt(d2) : 0.0f;
        float l    = __logf(nsI[v] + nsJ) + dist;
        l          = l > 0.0f ? l : 0.0f;
        accL      += m ? l * l : 0.0f;
        accC      += m ? 1.0f : 0.0f;
    }

    accL += __shfl_xor(accL, 1, 32);  accC += __shfl_xor(accC, 1, 32);
    accL += __shfl_xor(accL, 2, 32);  accC += __shfl_xor(accC, 2, 32);
    accL += __shfl_xor(accL, 4, 32);  accC += __shfl_xor(accC, 4, 32);
    accL += __shfl_xor(accL, 8, 32);  accC += __shfl_xor(accC, 8, 32);
    accL += __shfl_xor(accL, 16, 32); accC += __shfl_xor(accC, 16, 32);

    if (lane == 0 && (accL != 0.0f || accC != 0.0f)) {
        atomicAdd(loss, accL);
        atomicAdd(cnt,  accC);
    }
}

// ---- Kernel 4: out = loss / (2 * max(count, 1)) ----
__global__ void finalize_kernel(const float* __restrict__ loss,
                                const float* __restrict__ cnt,
                                float* __restrict__ out) {
    float c = *cnt;
    if (c < 1.0f) c = 1.0f;
    out[0] = *loss / (2.0f * c);
}

extern "C" void kernel_launch(void* const* d_in, const int* in_sizes, int n_in,
                              void* d_out, int out_size, void* d_ws, size_t ws_size,
                              hipStream_t stream) {
    const float*     score = (const float*)d_in[0];       // [4096,128] f32
    const long long* tgt   = (const long long*)d_in[1];   // [4096] int64
    float*           out   = (float*)d_out;               // scalar f32

    char* ws = (char*)d_ws;
    _Float16* Xh  = (_Float16*)ws;                        // 4096*128*2 = 1 MB
    float* mag    = (float*)(ws + (1u << 20));            // 16 KB
    float* neg    = (float*)(ws + (1u << 20) + B_ROWS * 4);
    float* loss   = (float*)(ws + (1u << 20) + 2 * B_ROWS * 4);
    float* cntf   = loss + 1;

    prep_kernel<<<B_ROWS / 8, 256, 0, stream>>>(score, Xh, mag, neg, loss, cntf);
    negsum_kernel<<<NTILE, 256, 0, stream>>>(Xh, mag, tgt, neg);
    dim3 grid2(NTILE, NTILE);
    pairloss_kernel<<<grid2, 32, 0, stream>>>(Xh, mag, tgt, neg, loss, cntf);
    finalize_kernel<<<1, 1, 0, stream>>>(loss, cntf, out);
}
